// GeodesicStateBlock_67534065762398
// MI455X (gfx1250) — matrix operand order repaired and checked by reading, not compiled, not run on hardware
//
#include <hip/hip_runtime.h>
#include <hip/hip_bf16.h>

#define B_   16
#define S_   2048
#define D_   512
#define TWO_D 1024
#define NWG  16
#define THREADS 128
#define PITCH 1032   // padded K pitch in bf16 elems (2064B rows: 16B aligned, bank-skewed)

typedef __attribute__((ext_vector_type(16))) __bf16 v16bf;
typedef __attribute__((ext_vector_type(8)))  __bf16 v8bf;
typedef __attribute__((ext_vector_type(4)))  __bf16 v4bf;
typedef __attribute__((ext_vector_type(8)))  float  v8f;

// Fast activations on the recurrence critical path: v_exp_f32 + v_rcp_f32 only,
// avoiding the IEEE v_div_scale/fixup sequence the compiler emits for '/'.
__device__ __forceinline__ float fast_sigmoid(float x) {
  return __builtin_amdgcn_rcpf(1.0f + __expf(-x));
}
__device__ __forceinline__ float fast_tanh(float x) {
  return 1.0f - 2.0f * __builtin_amdgcn_rcpf(1.0f + __expf(2.0f * x));
}

// Build a 16-element bf16 fragment from two contiguous 8-element (16B) LDS chunks.
__device__ __forceinline__ v16bf frag16(const __bf16* p0, const __bf16* p1) {
  v8bf lo = *(const v8bf*)p0;
  v8bf hi = *(const v8bf*)p1;
  v16bf r;
#pragma unroll
  for (int i = 0; i < 8; ++i) { r[i] = lo[i]; r[8 + i] = hi[i]; }
  return r;
}

// Device-wide sense/generation barrier across the 16 persistent workgroups.
__device__ __forceinline__ void wg_barrier(unsigned* cnt, unsigned* gen) {
  __syncthreads();
  if (threadIdx.x == 0) {
    __threadfence();
    unsigned g = __hip_atomic_load(gen, __ATOMIC_ACQUIRE, __HIP_MEMORY_SCOPE_AGENT);
    unsigned a = __hip_atomic_fetch_add(cnt, 1u, __ATOMIC_ACQ_REL, __HIP_MEMORY_SCOPE_AGENT);
    if (a == NWG - 1u) {
      __hip_atomic_store(cnt, 0u, __ATOMIC_RELAXED, __HIP_MEMORY_SCOPE_AGENT);
      __hip_atomic_fetch_add(gen, 1u, __ATOMIC_ACQ_REL, __HIP_MEMORY_SCOPE_AGENT);
    } else {
      while (__hip_atomic_load(gen, __ATOMIC_ACQUIRE, __HIP_MEMORY_SCOPE_AGENT) == g) {
        __builtin_amdgcn_s_sleep(2);
      }
    }
  }
  __syncthreads();
}

__global__ void __launch_bounds__(THREADS, 1)
gru_persistent(const float* __restrict__ points,
               const float* __restrict__ Wg,
               const float* __restrict__ bg,
               const float* __restrict__ Wc,
               const float* __restrict__ bc,
               float* __restrict__ out,
               __bf16* __restrict__ RH,      // [16][512] bf16 staging (r*h)
               float*  __restrict__ Hbuf,    // [16][512] f32 staging (h_new)
               unsigned* __restrict__ cnt,
               unsigned* __restrict__ gen)
{
  // ~238 KB LDS: legal on CDNA5 (320 KB/WGP); forces 1 WG per WGP -> all 16 resident.
  __shared__ __bf16 sA [B_][PITCH];   // [p_t | h or r*h], bf16
  __shared__ __bf16 sWg[64][PITCH];   // gate weight slice, column-major (n, k)
  __shared__ __bf16 sWc[32][PITCH];   // cand weight slice, column-major
  __shared__ float  sh [B_][40];      // local f32 h slice, cols [wg*32, wg*32+32)
  __shared__ float  red[4][16][16];   // K-split partial reduction for cand

  const int wg   = blockIdx.x;        // 0..15
  const int tid  = threadIdx.x;
  const int wave = tid >> 5;
  const int lane = tid & 31;
  const int l15  = lane & 15;         // A-row m / B-col n for this lane
  const int kh   = lane >> 4;         // lane half selects K region
  const int mB   = kh * 8;            // C/D row base for this lane

  // ---------------- one-time staging ----------------
  // Gate slice: rows 0..31 -> r cols [wg*32, wg*32+32); rows 32..63 -> matching z cols.
  for (int idx = tid; idx < 64 * TWO_D; idx += THREADS) {
    int nl = idx & 63, k = idx >> 6;
    int col = (nl < 32) ? (wg * 32 + nl) : (512 + wg * 32 + (nl - 32));
    sWg[nl][k] = (__bf16)Wg[k * TWO_D + col];
  }
  for (int idx = tid; idx < 32 * TWO_D; idx += THREADS) {
    int nl = idx & 31, k = idx >> 5;
    sWc[nl][k] = (__bf16)Wc[k * D_ + (wg * 32 + nl)];
  }
  // h0 = 0; load p_0 panel (vectorized: float4 in, v4bf out)
  for (int idx = tid; idx < B_ * D_ / 4; idx += THREADS) {
    int m = idx >> 7, c = (idx & 127) * 4;
    float4 pv = *(const float4*)&points[(size_t)m * S_ * D_ + c];
    v4bf o = { (__bf16)pv.x, (__bf16)pv.y, (__bf16)pv.z, (__bf16)pv.w };
    *(v4bf*)&sA[m][c] = o;
    v4bf zz = { (__bf16)0.f, (__bf16)0.f, (__bf16)0.f, (__bf16)0.f };
    *(v4bf*)&sA[m][512 + c] = zz;
  }
  for (int idx = tid; idx < B_ * 32; idx += THREADS) sh[idx >> 5][idx & 31] = 0.0f;
  __syncthreads();

  const float bias_g = bg[(wave < 2) ? (wg * 32 + wave * 16 + l15)
                                     : (512 + wg * 32 + (wave - 2) * 16 + l15)];
  const float bias_c = bc[wg * 32 + ((wave >= 2) ? (wave - 2) : 0) * 16 + l15];

  const __bf16* aBase  = &sA[l15][kh * 8];
  const __bf16* bgBase = &sWg[wave * 16 + l15][kh * 8];
  const __bf16* bcBase = &sWc[(wave >> 1) * 16 + l15][kh * 8];
  const int khalf = wave & 1;

  for (int t = 0; t < S_; ++t) {
    // -------- phase 1: gates tile = [p_t | h] @ Wg_slice (each wave: one 16x16 N-tile)
    v8f acc = {0.f,0.f,0.f,0.f,0.f,0.f,0.f,0.f};
#pragma unroll 4
    for (int k0 = 0; k0 < TWO_D; k0 += 32) {
      v16bf a = frag16(aBase  + k0, aBase  + k0 + 16);
      v16bf b = frag16(bgBase + k0, bgBase + k0 + 16);
      acc = __builtin_amdgcn_wmma_f32_16x16x32_bf16(false, a, false, b,
                                                    (short)0, acc, false, false);
    }
    float sg[8];
#pragma unroll
    for (int r = 0; r < 8; ++r) sg[r] = fast_sigmoid(acc[r] + bias_g);

    v8f z = {0.f,0.f,0.f,0.f,0.f,0.f,0.f,0.f};
    if (wave < 2) {                 // r-gate waves: publish r*h (bf16) to L2
      int col = wg * 32 + wave * 16 + l15;
      int lc  = wave * 16 + l15;
#pragma unroll
      for (int r = 0; r < 8; ++r)
        RH[(mB + r) * D_ + col] = (__bf16)(sg[r] * sh[mB + r][lc]);
    } else {                        // z-gate waves: keep z in registers
#pragma unroll
      for (int r = 0; r < 8; ++r) z[r] = sg[r];
    }

    // prefetch next timestep's points rows (global_prefetch_b8)
    if (t + 1 < S_) {
      int pm = tid >> 3, pc = (tid & 7) * 64;
      __builtin_prefetch(&points[(size_t)pm * S_ * D_ + (size_t)(t + 1) * D_ + pc], 0, 0);
    }

    wg_barrier(cnt, gen);           // all WGs' r*h visible

    // -------- phase 2 prep: A2 = [p_t | r*h]  (16B vector copies, 8 iters/thread)
    for (int idx = tid; idx < B_ * D_ / 8; idx += THREADS) {
      int m = idx >> 6, c = (idx & 63) * 8;
      *(v8bf*)&sA[m][512 + c] = *(const v8bf*)&RH[m * D_ + c];
    }
    __syncthreads();

    // -------- phase 2: cand tile partials (2 N-tiles x K-halves over 4 waves)
    v8f acc2 = {0.f,0.f,0.f,0.f,0.f,0.f,0.f,0.f};
    const int kbeg = khalf * 512;
#pragma unroll 4
    for (int k0 = kbeg; k0 < kbeg + 512; k0 += 32) {
      v16bf a = frag16(aBase  + k0, aBase  + k0 + 16);
      v16bf b = frag16(bcBase + k0, bcBase + k0 + 16);
      acc2 = __builtin_amdgcn_wmma_f32_16x16x32_bf16(false, a, false, b,
                                                     (short)0, acc2, false, false);
    }
#pragma unroll
    for (int r = 0; r < 8; ++r) red[wave][mB + r][l15] = acc2[r];
    __syncthreads();

    // waves 2,3 (holders of z) reduce, activate, and update h
    if (wave >= 2) {
      const int t2 = wave - 2;
#pragma unroll
      for (int r = 0; r < 8; ++r) {
        float c  = red[2 * t2][mB + r][l15] + red[2 * t2 + 1][mB + r][l15] + bias_c;
        c = fast_tanh(c);
        float h  = sh[mB + r][t2 * 16 + l15];
        float hn = (1.0f - z[r]) * h + z[r] * c;
        int col  = wg * 32 + t2 * 16 + l15;
        Hbuf[(mB + r) * D_ + col] = hn;
        out[(size_t)(mB + r) * S_ * D_ + (size_t)t * D_ + col] = hn;  // state (==deformed)
      }
    }

    wg_barrier(cnt, gen);           // all WGs' h_new visible

    // -------- phase 3: refresh p_{t+1} and h in LDS (vectorized)
    if (t + 1 < S_) {
      for (int idx = tid; idx < B_ * D_ / 4; idx += THREADS) {
        int m = idx >> 7, c = (idx & 127) * 4;
        float4 pv = *(const float4*)&points[(size_t)m * S_ * D_ + (size_t)(t + 1) * D_ + c];
        v4bf o = { (__bf16)pv.x, (__bf16)pv.y, (__bf16)pv.z, (__bf16)pv.w };
        *(v4bf*)&sA[m][c] = o;
      }
    }
    for (int idx = tid; idx < B_ * D_ / 4; idx += THREADS) {
      int m = idx >> 7, c = (idx & 127) * 4;
      float4 hv = *(const float4*)&Hbuf[m * D_ + c];
      v4bf o = { (__bf16)hv.x, (__bf16)hv.y, (__bf16)hv.z, (__bf16)hv.w };
      *(v4bf*)&sA[m][512 + c] = o;
      int lc = c - wg * 32;
      if (lc >= 0 && lc < 32) {
        sh[m][lc]     = hv.x;  sh[m][lc + 1] = hv.y;
        sh[m][lc + 2] = hv.z;  sh[m][lc + 3] = hv.w;
      }
    }
    __syncthreads();
  }
}

// In-place LayerNorm: one wave per 512-elem row. 256 MB total traffic -> ~11us at 23.3 TB/s.
__global__ void __launch_bounds__(128)
layernorm_inplace(float* __restrict__ x, const float* __restrict__ gamma,
                  const float* __restrict__ beta)
{
  int row  = blockIdx.x * 4 + (threadIdx.x >> 5);
  int lane = threadIdx.x & 31;
  if (row >= B_ * S_) return;
  float* p = x + (size_t)row * D_;
  float v[16];
  float s = 0.f;
#pragma unroll
  for (int i = 0; i < 16; ++i) { v[i] = p[lane + 32 * i]; s += v[i]; }
#pragma unroll
  for (int o = 16; o > 0; o >>= 1) s += __shfl_xor(s, o, 32);
  float mu = s * (1.0f / D_);
  float q = 0.f;
#pragma unroll
  for (int i = 0; i < 16; ++i) { float d = v[i] - mu; q += d * d; }
#pragma unroll
  for (int o = 16; o > 0; o >>= 1) q += __shfl_xor(q, o, 32);
  float rstd = rsqrtf(q * (1.0f / D_) + 1e-5f);
#pragma unroll
  for (int i = 0; i < 16; ++i) {
    int c = lane + 32 * i;
    p[c] = (v[i] - mu) * rstd * gamma[c] + beta[c];
  }
}

__global__ void init_sync(unsigned* p) {
  if (threadIdx.x < 2) p[threadIdx.x] = 0u;   // re-zero barrier state every call
}

extern "C" void kernel_launch(void* const* d_in, const int* in_sizes, int n_in,
                              void* d_out, int out_size, void* d_ws, size_t ws_size,
                              hipStream_t stream) {
  const float* points = (const float*)d_in[0];
  const float* Wg     = (const float*)d_in[1];
  const float* bg     = (const float*)d_in[2];
  const float* Wc     = (const float*)d_in[3];
  const float* bc     = (const float*)d_in[4];
  const float* gamma  = (const float*)d_in[5];
  const float* beta   = (const float*)d_in[6];
  float* out = (float*)d_out;

  char* ws = (char*)d_ws;
  __bf16*   RH   = (__bf16*)ws;                    // 16 KB: r*h staging
  float*    Hbuf = (float*)(ws + 16 * 1024);       // 32 KB: h_new staging
  unsigned* sync = (unsigned*)(ws + 64 * 1024);    // barrier counter + generation

  init_sync<<<1, 32, 0, stream>>>(sync);
  gru_persistent<<<NWG, THREADS, 0, stream>>>(points, Wg, bg, Wc, bc, out,
                                              RH, Hbuf, sync, sync + 1);
  layernorm_inplace<<<(B_ * S_) / 4, 128, 0, stream>>>(out, gamma, beta);
}
